// MultiHeadAttention_70437463654510
// MI455X (gfx1250) — compile-verified, matrix operand order checked by reading, not run on hardware
//
#include <hip/hip_runtime.h>

// MHA forward for MI455X (gfx1250, wave32, WMMA).
// B=2, T=2048, D=768, H=12, HS=64.
// ~45 GFLOP of GEMM work vs ~150MB of traffic @23.3TB/s -> matrix-pipe bound;
// use v_wmma_f32_16x16x32_bf16 (K=32/instr, ~8x the K=4 f32 WMMA rate) with
// f32 accumulation. All fp32->bf16 conversion is done ONCE in a bandwidth-
// bound pass so the GEMM inner loops are pure global_load_b128 + WMMA.

#define BB 2
#define TT 2048
#define DD 768
#define HH 12
#define HS 64
#define BT (BB*TT)

typedef __attribute__((ext_vector_type(16))) __bf16 v16bf;
typedef __attribute__((ext_vector_type(8)))  __bf16 v8bf;
typedef __attribute__((ext_vector_type(8)))  float  v8f;
typedef __attribute__((ext_vector_type(4)))  float  v4f;

// Native f32 -> bf16 (compiler picks the hw cvt; gfx1250 has true bf16 VALU).
__device__ __forceinline__ __bf16 f2bf(float f) { return (__bf16)f; }

// D = A(16x32 bf16) x B(32x16 bf16) + C(16x16 f32)
__device__ __forceinline__ v8f wmma_bf16(v16bf a, v16bf b, v8f c) {
    return __builtin_amdgcn_wmma_f32_16x16x32_bf16(
        /*neg_a=*/false, a, /*neg_b=*/false, b,
        /*c_mod=*/(short)0, c, /*reuse_a=*/false, /*reuse_b=*/false);
}

// Load a 16x32 bf16 WMMA operand from a row-major bf16 matrix.
// ISA A-layout (B-layout identical with N<->M roles): lane (half,r), r=lane&15,
// half=lane>>4 holds row r, K elems {half*8..half*8+7, 16+half*8..16+half*8+7}.
// base points at element [row0, k0]; ld = row stride in elements.
// -> two contiguous 16B loads per lane (global_load_b128 / ds_load_b128).
__device__ __forceinline__ v16bf load_tile_bf16(const __bf16* base, int ld, int r, int half) {
    const __bf16* p0 = base + (size_t)r * ld + half * 8;
    union { v16bf v; struct { v8bf lo, hi; } s; } u;
    u.s.lo = *(const v8bf*)p0;
    u.s.hi = *(const v8bf*)(p0 + 16);
    return u.v;
}

// ---------------------------------------------------------------------------
// Kernel 0a: zero the attn-mean region of d_out (accumulated via f32 atomics).
__global__ void zero_mean_kernel(float* __restrict__ p, size_t n4) {
    size_t i = (size_t)blockIdx.x * blockDim.x + threadIdx.x;
    if (i < n4) {
        v4f z = {0.f, 0.f, 0.f, 0.f};
        *(v4f*)(p + i * 4) = z;
    }
}

// Kernel 0b: bulk f32 -> bf16 conversion (bandwidth-bound, runs once).
// 8 elements/thread: 2x b128 loads -> 1x b128 store.
__global__ void cvt_bf16_kernel(const float* __restrict__ src,
                                __bf16* __restrict__ dst, unsigned n8) {
    unsigned i = blockIdx.x * blockDim.x + threadIdx.x;
    if (i < n8) {
        const v4f x0 = *(const v4f*)(src + (size_t)i * 8);
        const v4f x1 = *(const v4f*)(src + (size_t)i * 8 + 4);
        v8bf o;
#pragma unroll
        for (int j = 0; j < 4; ++j) {
            o[j]     = f2bf(x0[j]);
            o[4 + j] = f2bf(x1[j]);
        }
        *(v8bf*)(dst + (size_t)i * 8) = o;
    }
}

// ---------------------------------------------------------------------------
// Kernel 1: Q/K/V projections.  out = x @ W^T + b  (M=BT, N=D, K=D), all-bf16
// operands. z = blockIdx.z selects {Q,K,V}. Each wave computes a 64(M)x64(N)
// tile: per K-step of 32: 4 A-operands + 4 B-operands + 16 WMMAs
// (1 b128 load per WMMA -> matrix-pipe dense inner loop).
// Q is pre-scaled by 1/sqrt(HS)=0.125 (folds the attention score scale).
// Q,K stored [B,H,T,HS] bf16; V stored transposed [B,H,HS,T] bf16 ->
// contiguous 16B stores AND contiguous B-operand loads for the P@V GEMM.
__global__ __launch_bounds__(128) void qkv_proj_kernel(
    const __bf16* __restrict__ Xq, const __bf16* __restrict__ Xk,
    const __bf16* __restrict__ Xv,
    const __bf16* __restrict__ Wqb, const __bf16* __restrict__ Wkb,
    const __bf16* __restrict__ Wvb,
    const float* __restrict__ bq, const float* __restrict__ bk,
    const float* __restrict__ bv,
    __bf16* __restrict__ Qb, __bf16* __restrict__ Kb, __bf16* __restrict__ Vt)
{
    const int lane = threadIdx.x & 31;
    const int wave = threadIdx.x >> 5;
    const int r = lane & 15, half = lane >> 4;
    const int z  = blockIdx.z;
    const int m0 = blockIdx.y * 64;                 // 64 row blocks of 64
    const int n0 = (blockIdx.x * 4 + wave) * 64;    // 12 col blocks of 64

    const __bf16* X    = (z == 0) ? Xq  : (z == 1) ? Xk  : Xv;
    const __bf16* W    = (z == 0) ? Wqb : (z == 1) ? Wkb : Wvb;
    const float*  bias = (z == 0) ? bq  : (z == 1) ? bk  : bv;

    v8f acc[4][4] = {};
    for (int k0 = 0; k0 < DD; k0 += 32) {
        v16bf a[4];
#pragma unroll
        for (int mi = 0; mi < 4; ++mi)
            a[mi] = load_tile_bf16(X + (size_t)(m0 + mi * 16) * DD + k0, DD, r, half);
#pragma unroll
        for (int ni = 0; ni < 4; ++ni) {
            v16bf bt = load_tile_bf16(W + (size_t)(n0 + ni * 16) * DD + k0, DD, r, half);
#pragma unroll
            for (int mi = 0; mi < 4; ++mi)
                acc[mi][ni] = wmma_bf16(a[mi], bt, acc[mi][ni]);
        }
    }

    const float qscale = (z == 0) ? 0.125f : 1.0f;
#pragma unroll
    for (int mi = 0; mi < 4; ++mi) {
#pragma unroll
        for (int ni = 0; ni < 4; ++ni) {
            const int n = n0 + ni * 16 + r;
            const float bias_v = bias[n];
            const int h = n / HS, hs = n % HS;
            const int mbase = m0 + mi * 16 + half * 8; // D-layout: elem j -> row mbase+j
            if (z != 2) {
                __bf16* dst = (z == 0) ? Qb : Kb;
#pragma unroll
                for (int j = 0; j < 8; ++j) {
                    const int m = mbase + j;
                    const int b = m >> 11, t = m & (TT - 1);
                    float val = (acc[mi][ni][j] + bias_v) * qscale;
                    dst[(((size_t)(b * HH + h)) * TT + t) * HS + hs] = f2bf(val);
                }
            } else {
                // transposed V: consecutive j -> consecutive t -> 16B store
                const int b = mbase >> 11, t0 = mbase & (TT - 1);
                __bf16 tmp[8];
#pragma unroll
                for (int j = 0; j < 8; ++j) tmp[j] = f2bf(acc[mi][ni][j] + bias_v);
                *(v8bf*)(Vt + (((size_t)(b * HH + h)) * HS + hs) * TT + t0) = *(v8bf*)tmp;
            }
        }
    }
}

// ---------------------------------------------------------------------------
// Kernel 2: attention. One wave owns 16 query rows of one (b,h).
// Pass 1: stream 128 key tiles, online softmax (m,l) per row, merged across
//         the 16-lane half-group with wave32 shfl_xor. (exp is TRANS and
//         co-executes with the XDL WMMA pipe.)
// Pass 2: recompute scores (WMMA-cheap), P = exp(s-m)/l, atomically add P/H
//         into attn-mean (global_atomic_add_f32), transpose P via a per-wave
//         LDS tile into A-layout, O += P @ V with transposed-V B operands.
__global__ __launch_bounds__(128) void attention_kernel(
    const __bf16* __restrict__ Qb, const __bf16* __restrict__ Kb,
    const __bf16* __restrict__ Vt, __bf16* __restrict__ Ob,
    float* __restrict__ attn_mean)
{
    __shared__ __align__(16) __bf16 stag[4][16 * 32];  // per-wave P staging tile

    const int lane = threadIdx.x & 31, wave = threadIdx.x >> 5;
    const int r = lane & 15, half = lane >> 4;
    const int b = blockIdx.z, h = blockIdx.y;
    const int m0 = (blockIdx.x * 4 + wave) * 16;

    const __bf16* Qh = Qb + (size_t)(b * HH + h) * TT * HS;
    const __bf16* Kh = Kb + (size_t)(b * HH + h) * TT * HS;
    const __bf16* Vh = Vt + (size_t)(b * HH + h) * HS * TT;

    // Q A-operands held in registers across both passes (already scale-folded)
    const v16bf aq0 = load_tile_bf16(Qh + (size_t)m0 * HS + 0,  HS, r, half);
    const v16bf aq1 = load_tile_bf16(Qh + (size_t)m0 * HS + 32, HS, r, half);

    // ---- pass 1: per-lane online (max,sum); lane (half,r) sees col n0+r of
    // rows half*8+0..7 per tile.
    float mrow[8], lrow[8];
#pragma unroll
    for (int j = 0; j < 8; ++j) { mrow[j] = -1e30f; lrow[j] = 0.f; }

    for (int n0 = 0; n0 < TT; n0 += 16) {
        v16bf bk0 = load_tile_bf16(Kh + (size_t)n0 * HS + 0,  HS, r, half);
        v16bf bk1 = load_tile_bf16(Kh + (size_t)n0 * HS + 32, HS, r, half);
        v8f s = {};
        s = wmma_bf16(aq0, bk0, s);
        s = wmma_bf16(aq1, bk1, s);
#pragma unroll
        for (int j = 0; j < 8; ++j) {
            float sv = s[j];
            float mn = fmaxf(mrow[j], sv);
            lrow[j] = lrow[j] * __expf(mrow[j] - mn) + __expf(sv - mn);
            mrow[j] = mn;
        }
    }
    // merge (m,l) across the 16 lanes of each half-group (rows differ per half)
#pragma unroll
    for (int mask = 1; mask < 16; mask <<= 1) {
#pragma unroll
        for (int j = 0; j < 8; ++j) {
            float mo = __shfl_xor(mrow[j], mask, 16);
            float lo = __shfl_xor(lrow[j], mask, 16);
            float mn = fmaxf(mrow[j], mo);
            lrow[j] = lrow[j] * __expf(mrow[j] - mn) + lo * __expf(mo - mn);
            mrow[j] = mn;
        }
    }
    float invl[8];
#pragma unroll
    for (int j = 0; j < 8; ++j) invl[j] = 1.0f / lrow[j];

    // ---- pass 2: P (normalized) -> mean atomics + LDS transpose -> O += P@V
    v8f o[4] = {};
    __bf16* st = stag[wave];
    const float hinv = 1.0f / (float)HH;

    for (int kb = 0; kb < TT; kb += 32) {
#pragma unroll
        for (int sidx = 0; sidx < 2; ++sidx) {
            const int n0 = kb + sidx * 16;
            v16bf bk0 = load_tile_bf16(Kh + (size_t)n0 * HS + 0,  HS, r, half);
            v16bf bk1 = load_tile_bf16(Kh + (size_t)n0 * HS + 32, HS, r, half);
            v8f s = {};
            s = wmma_bf16(aq0, bk0, s);
            s = wmma_bf16(aq1, bk1, s);
            const size_t mbase = (size_t)(b * TT + m0 + half * 8) * TT + n0 + r;
#pragma unroll
            for (int j = 0; j < 8; ++j) {
                float p = __expf(s[j] - mrow[j]) * invl[j];  // normalized prob
                __hip_atomic_fetch_add(attn_mean + mbase + (size_t)j * TT,
                                       p * hinv, __ATOMIC_RELAXED,
                                       __HIP_MEMORY_SCOPE_AGENT);
                // row-major P tile in LDS: row = half*8+j, col = sidx*16+r
                st[(half * 8 + j) * 32 + sidx * 16 + r] = f2bf(p);
            }
        }
        // intra-wave LDS RAW: order the scattered b16 stores vs the b128 loads
        asm volatile("s_wait_dscnt 0" ::: "memory");
        v16bf ap = load_tile_bf16(st, 32, r, half);  // P in A-layout (16x32)
#pragma unroll
        for (int a = 0; a < 4; ++a) {
            // B operand: col n=a*16+r of V tile = row of Vt -> contiguous
            v16bf bv = load_tile_bf16(Vh + (size_t)(a * 16) * TT + kb, TT, r, half);
            o[a] = wmma_bf16(ap, bv, o[a]);
        }
    }

    // O already normalized (P was divided by l). Store bf16 [BT, D].
#pragma unroll
    for (int a = 0; a < 4; ++a) {
        const int hs = a * 16 + r;
#pragma unroll
        for (int j = 0; j < 8; ++j) {
            const int t = m0 + half * 8 + j;
            Ob[(size_t)(b * TT + t) * DD + h * HS + hs] = f2bf(o[a][j]);
        }
    }
}

// ---------------------------------------------------------------------------
// Kernel 3: output projection.  out = O @ Wo^T + bo, 64x64 tile per wave,
// all-bf16 operands, f32 result straight to d_out.
__global__ __launch_bounds__(128) void out_proj_kernel(
    const __bf16* __restrict__ Ob, const __bf16* __restrict__ Wob,
    const float* __restrict__ bo, float* __restrict__ out)
{
    const int lane = threadIdx.x & 31;
    const int wave = threadIdx.x >> 5;
    const int r = lane & 15, half = lane >> 4;
    const int m0 = blockIdx.y * 64;
    const int n0 = (blockIdx.x * 4 + wave) * 64;

    v8f acc[4][4] = {};
    for (int k0 = 0; k0 < DD; k0 += 32) {
        v16bf a[4];
#pragma unroll
        for (int mi = 0; mi < 4; ++mi)
            a[mi] = load_tile_bf16(Ob + (size_t)(m0 + mi * 16) * DD + k0, DD, r, half);
#pragma unroll
        for (int ni = 0; ni < 4; ++ni) {
            v16bf bt = load_tile_bf16(Wob + (size_t)(n0 + ni * 16) * DD + k0, DD, r, half);
#pragma unroll
            for (int mi = 0; mi < 4; ++mi)
                acc[mi][ni] = wmma_bf16(a[mi], bt, acc[mi][ni]);
        }
    }
#pragma unroll
    for (int mi = 0; mi < 4; ++mi) {
#pragma unroll
        for (int ni = 0; ni < 4; ++ni) {
            const int n = n0 + ni * 16 + r;
            const float bias_v = bo[n];
            const int mbase = m0 + mi * 16 + half * 8;
#pragma unroll
            for (int j = 0; j < 8; ++j)
                out[(size_t)(mbase + j) * DD + n] = acc[mi][ni][j] + bias_v;
        }
    }
}

// ---------------------------------------------------------------------------
extern "C" void kernel_launch(void* const* d_in, const int* in_sizes, int n_in,
                              void* d_out, int out_size, void* d_ws, size_t ws_size,
                              hipStream_t stream) {
    (void)in_sizes; (void)n_in; (void)out_size; (void)ws_size;

    const float* q_in = (const float*)d_in[0];
    const float* k_in = (const float*)d_in[1];
    const float* v_in = (const float*)d_in[2];
    const float* Wq   = (const float*)d_in[3];
    const float* bq   = (const float*)d_in[4];
    const float* Wk   = (const float*)d_in[5];
    const float* bk   = (const float*)d_in[6];
    const float* Wv   = (const float*)d_in[7];
    const float* bv   = (const float*)d_in[8];
    const float* Wo   = (const float*)d_in[9];
    const float* bo   = (const float*)d_in[10];

    float* out  = (float*)d_out;                       // [B,T,D] f32
    float* mean = out + (size_t)BB * TT * DD;          // [B,T,T] f32

    // workspace layout (bf16 elements), ~49 MB total:
    const size_t nAct = (size_t)BT * DD;   // 3,145,728
    const size_t nW   = (size_t)DD * DD;   //   589,824
    __bf16* Xq  = (__bf16*)d_ws;           // bf16 copies of activations
    __bf16* Xk  = Xq  + nAct;
    __bf16* Xv  = Xk  + nAct;
    __bf16* Wqb = Xv  + nAct;              // bf16 copies of weights
    __bf16* Wkb = Wqb + nW;
    __bf16* Wvb = Wkb + nW;
    __bf16* Wob = Wvb + nW;
    __bf16* Qb  = Wob + nW;                // [B,H,T,HS]
    __bf16* Kb  = Qb  + nAct;              // [B,H,T,HS]
    __bf16* Vt  = Kb  + nAct;              // [B,H,HS,T] (transposed)
    __bf16* Ob  = Vt  + nAct;              // [BT, D]

    // 0a) zero the attn-mean accumulation region
    const size_t meanN4 = (size_t)BB * TT * TT / 4;    // 2,097,152 float4s
    zero_mean_kernel<<<dim3((unsigned)(meanN4 / 256)), 256, 0, stream>>>(mean, meanN4);

    // 0b) one-shot f32 -> bf16 conversions (bandwidth-bound)
    const unsigned actN8 = (unsigned)(nAct / 8), wN8 = (unsigned)(nW / 8);
    cvt_bf16_kernel<<<actN8 / 256, 256, 0, stream>>>(q_in, Xq, actN8);
    cvt_bf16_kernel<<<actN8 / 256, 256, 0, stream>>>(k_in, Xk, actN8);
    cvt_bf16_kernel<<<actN8 / 256, 256, 0, stream>>>(v_in, Xv, actN8);
    cvt_bf16_kernel<<<wN8   / 256, 256, 0, stream>>>(Wq,   Wqb, wN8);
    cvt_bf16_kernel<<<wN8   / 256, 256, 0, stream>>>(Wk,   Wkb, wN8);
    cvt_bf16_kernel<<<wN8   / 256, 256, 0, stream>>>(Wv,   Wvb, wN8);
    cvt_bf16_kernel<<<wN8   / 256, 256, 0, stream>>>(Wo,   Wob, wN8);

    // 1) QKV projections: grid (12 n-tiles / 4 waves, 64 m-tiles, 3 matrices)
    qkv_proj_kernel<<<dim3(3, BT / 64, 3), 128, 0, stream>>>(
        Xq, Xk, Xv, Wqb, Wkb, Wvb, bq, bk, bv, Qb, Kb, Vt);

    // 2) attention: grid (T/64, H, B), 4 waves/block, 16 q-rows/wave
    attention_kernel<<<dim3(TT / 64, HH, BB), 128, 0, stream>>>(Qb, Kb, Vt, Ob, mean);

    // 3) output projection: grid (12 n-tiles / 4 waves, 64 m-tiles)
    out_proj_kernel<<<dim3(3, BT / 64), 128, 0, stream>>>(Ob, Wob, bo, out);
}
